// ATTGCN_53824530153631
// MI455X (gfx1250) — matrix operand order repaired and checked by reading, not compiled, not run on hardware
//
#include <hip/hip_runtime.h>
#include <hip/hip_bf16.h>

#define DEV __device__ __forceinline__

constexpr int Nn = 50000;
constexpr int Ee = 640000;
constexpr int EP = Ee + Nn;     // edges + self loops
constexpr int Hh = 128;
constexpr int Gg = 256;
constexpr int Cc = 10;
constexpr float SLOPE = 0.2f;

typedef _Float16 h16v __attribute__((ext_vector_type(16)));
typedef float    f8v  __attribute__((ext_vector_type(8)));

DEV int   f2oi(float f) { int i = __float_as_int(f); return i >= 0 ? i : (i ^ 0x7fffffff); }
DEV float oi2f(int i)   { return __int_as_float(i >= 0 ? i : (i ^ 0x7fffffff)); }
DEV void  atomAddF(float* p, float v) { unsafeAtomicAdd(p, v); }   // native global_atomic_add_f32
DEV void  atomMaxI(int* p, int v) { atomicMax(p, v); }             // native global_atomic_max_i32

// Wt[j*128 + k] = W[k*128 + j]  (transpose so B loads are K-contiguous per column)
__global__ void cvtW_kernel(const float* __restrict__ W, _Float16* __restrict__ Wt) {
    int t = blockIdx.x * blockDim.x + threadIdx.x;   // 0..16383
    int k = t >> 7, j = t & 127;
    Wt[j * Hh + k] = (_Float16)W[k * Hh + j];
}

// ---------------- WMMA GEMM: H[N,128] = f16(Ain[N,128]) @ W[128,128] ----------------
// One wave per 16-row tile; A converted f32->f16 in-register; sweep 8 col tiles x 4 K-chunks.
__global__ void gemm_wmma_kernel(const float* Ain,
                                 const _Float16* __restrict__ Wt,
                                 float* Hout) {
    int wave = (blockIdx.x * blockDim.x + threadIdx.x) >> 5;
    if (wave >= Nn / 16) return;                 // wave-uniform exit, EXEC stays all-1s inside
    int lane = threadIdx.x & 31;
    int row  = lane & 15;
    int half = lane >> 4;
    int rowBase = wave * 16;

    const float* arow = Ain + (size_t)(rowBase + row) * Hh;
    h16v a[4];
#pragma unroll
    for (int kc = 0; kc < 4; ++kc) {
        // lane covers K = kc*32 + 16h .. +15
        float4 f0 = *(const float4*)(arow + kc * 32 + 16 * half);
        float4 f1 = *(const float4*)(arow + kc * 32 + 16 * half + 4);
        float4 f2 = *(const float4*)(arow + kc * 32 + 16 * half + 8);
        float4 f3 = *(const float4*)(arow + kc * 32 + 16 * half + 12);
        h16v v;
        v[0]  = (_Float16)f0.x; v[1]  = (_Float16)f0.y; v[2]  = (_Float16)f0.z; v[3]  = (_Float16)f0.w;
        v[4]  = (_Float16)f1.x; v[5]  = (_Float16)f1.y; v[6]  = (_Float16)f1.z; v[7]  = (_Float16)f1.w;
        v[8]  = (_Float16)f2.x; v[9]  = (_Float16)f2.y; v[10] = (_Float16)f2.z; v[11] = (_Float16)f2.w;
        v[12] = (_Float16)f3.x; v[13] = (_Float16)f3.y; v[14] = (_Float16)f3.z; v[15] = (_Float16)f3.w;
        a[kc] = v;
    }

#pragma unroll
    for (int ct = 0; ct < 8; ++ct) {
        int col = ct * 16 + (lane & 15);
        f8v c = {};
#pragma unroll
        for (int kc = 0; kc < 4; ++kc) {
            h16v b = *(const h16v*)(Wt + (size_t)col * Hh + kc * 32 + 16 * half);
            c = __builtin_amdgcn_wmma_f32_16x16x32_f16(false, a[kc], false, b,
                                                       (short)0, c, false, false);
        }
        float* outp = Hout + (size_t)(rowBase + 8 * half) * Hh + ct * 16 + (lane & 15);
#pragma unroll
        for (int r = 0; r < 8; ++r) outp[(size_t)r * Hh] = c[r];
    }
}

// ---------------- per-node attention terms ----------------
__global__ void node_attn_kernel(const float* __restrict__ H,
                                 const float* __restrict__ as, const float* __restrict__ ad,
                                 float* __restrict__ es, float* __restrict__ ed) {
    int n = blockIdx.x * blockDim.x + threadIdx.x;
    if (n >= Nn) return;
    const float4* hp  = (const float4*)(H + (size_t)n * Hh);
    const float4* asp = (const float4*)as;
    const float4* adp = (const float4*)ad;
    float s = 0.f, d = 0.f;
#pragma unroll 8
    for (int i = 0; i < 32; ++i) {
        float4 h = hp[i], a = asp[i], b = adp[i];
        s += h.x * a.x + h.y * a.y + h.z * a.z + h.w * a.w;
        d += h.x * b.x + h.y * b.y + h.z * b.z + h.w * b.w;
    }
    es[n] = s; ed[n] = d;
}

__global__ void init_node_kernel(int* __restrict__ mInt, float* __restrict__ z) {
    int n = blockIdx.x * blockDim.x + threadIdx.x;
    if (n < Nn) { mInt[n] = (int)0x80000000; z[n] = 0.f; }
}

// ---------------- edge passes ----------------
__global__ void edge_max_kernel(const int* __restrict__ esrc, const int* __restrict__ edst,
                                const float* __restrict__ es, const float* __restrict__ ed,
                                float* __restrict__ ew, int* __restrict__ mInt) {
    int e = blockIdx.x * blockDim.x + threadIdx.x;
    if (e >= EP) return;
    int s = (e < Ee) ? esrc[e] : (e - Ee);
    int d = (e < Ee) ? edst[e] : (e - Ee);
    float v = es[s] + ed[d];
    v = (v > 0.f) ? v : SLOPE * v;               // leaky_relu
    ew[e] = v;
    atomMaxI(&mInt[d], f2oi(v));
}

__global__ void edge_exp_kernel(const int* __restrict__ edst,
                                float* __restrict__ ew, const int* __restrict__ mInt,
                                float* __restrict__ z) {
    int e = blockIdx.x * blockDim.x + threadIdx.x;
    if (e >= EP) return;
    int d = (e < Ee) ? edst[e] : (e - Ee);
    float w = expf(ew[e] - oi2f(mInt[d]));
    ew[e] = w;
    atomAddF(&z[d], w);
}

// one wave per edge; each lane handles 4 features
__global__ void edge_scatter_kernel(const int* __restrict__ esrc, const int* __restrict__ edst,
                                    const float* __restrict__ ew, const float* __restrict__ z,
                                    const float* __restrict__ H, float* __restrict__ Out) {
    int gid = blockIdx.x * blockDim.x + threadIdx.x;
    int e = gid >> 5;
    if (e >= EP) return;
    int lane = gid & 31;
    int s = (e < Ee) ? esrc[e] : (e - Ee);
    int d = (e < Ee) ? edst[e] : (e - Ee);
    float alpha = ew[e] / (z[d] + 1e-16f);
    float4 h = *(const float4*)(H + (size_t)s * Hh + lane * 4);
    float* op = Out + (size_t)d * Hh + lane * 4;
    atomAddF(op + 0, alpha * h.x);
    atomAddF(op + 1, alpha * h.y);
    atomAddF(op + 2, alpha * h.z);
    atomAddF(op + 3, alpha * h.w);
}

__global__ void bias_relu_kernel(float* __restrict__ Out, const float* __restrict__ bc) {
    int i = blockIdx.x * blockDim.x + threadIdx.x;
    if (i >= Nn * Hh) return;
    float v = Out[i] + bc[i & 127];
    Out[i] = fmaxf(v, 0.f);
}

// ---------------- global max pool ----------------
__global__ void init_pool_kernel(int* __restrict__ gInt) {
    int i = blockIdx.x * blockDim.x + threadIdx.x;
    if (i < Gg * Hh) gInt[i] = (int)0x80000000;
}

__global__ void pool_max_kernel(const float* __restrict__ H, const int* __restrict__ batch,
                                int* __restrict__ gInt) {
    int i = blockIdx.x * blockDim.x + threadIdx.x;
    if (i >= Nn * Hh) return;
    int n = i >> 7, j = i & 127;
    atomMaxI(&gInt[batch[n] * Hh + j], f2oi(H[i]));
}

// ---------------- MLP head + log_softmax (tiny: 256 rows) ----------------
__global__ void mlp_kernel(const int* __restrict__ gInt,
                           const float* __restrict__ W1, const float* __restrict__ b1,
                           const float* __restrict__ W2, const float* __restrict__ b2,
                           float* __restrict__ out) {
    __shared__ float gs[Hh];
    __shared__ float hid[Hh];
    __shared__ float logits[Cc];
    int g = blockIdx.x, t = threadIdx.x;
    gs[t] = oi2f(gInt[g * Hh + t]);
    __syncthreads();
    float acc = b1[t];
#pragma unroll 8
    for (int k = 0; k < Hh; ++k) acc += gs[k] * W1[k * Hh + t];
    hid[t] = fmaxf(acc, 0.f);
    __syncthreads();
    if (t < Cc) {
        float a2 = b2[t];
#pragma unroll 8
        for (int k = 0; k < Hh; ++k) a2 += hid[k] * W2[k * Cc + t];
        logits[t] = a2;
    }
    __syncthreads();
    if (t < Cc) {
        float m = -3.4e38f;
#pragma unroll
        for (int c = 0; c < Cc; ++c) m = fmaxf(m, logits[c]);
        float s = 0.f;
#pragma unroll
        for (int c = 0; c < Cc; ++c) s += expf(logits[c] - m);
        out[g * Cc + t] = logits[t] - m - logf(s);
    }
}

// ---------------- host launcher ----------------
extern "C" void kernel_launch(void* const* d_in, const int* in_sizes, int n_in,
                              void* d_out, int out_size, void* d_ws, size_t ws_size,
                              hipStream_t stream) {
    const float* x     = (const float*)d_in[0];
    const int*   ei    = (const int*)d_in[1];
    const int*   batch = (const int*)d_in[2];
    const float* Wc    = (const float*)d_in[3];
    const float* a_src = (const float*)d_in[4];
    const float* a_dst = (const float*)d_in[5];
    const float* bc    = (const float*)d_in[6];
    const float* W1    = (const float*)d_in[7];
    const float* b1    = (const float*)d_in[8];
    const float* W2    = (const float*)d_in[9];
    const float* b2    = (const float*)d_in[10];
    const int* esrc = ei;
    const int* edst = ei + Ee;

    // workspace carve (256B aligned)
    char* base = (char*)d_ws;
    size_t off = 0;
    auto carve = [&](size_t bytes) -> void* {
        void* p = base + off;
        off = (off + bytes + 255) & ~(size_t)255;
        return p;
    };
    float*    buf0 = (float*)carve((size_t)Nn * Hh * 4);
    float*    buf1 = (float*)carve((size_t)Nn * Hh * 4);
    float*    buf2 = (float*)carve((size_t)Nn * Hh * 4);
    _Float16* wt16 = (_Float16*)carve((size_t)Hh * Hh * 2);
    float*    es   = (float*)carve((size_t)Nn * 4);
    float*    ed   = (float*)carve((size_t)Nn * 4);
    int*      mInt = (int*)carve((size_t)Nn * 4);
    float*    z    = (float*)carve((size_t)Nn * 4);
    float*    ew   = (float*)carve((size_t)EP * 4);
    int*      gInt = (int*)carve((size_t)Gg * Hh * 4);

    const int B = 256;
    const int gElem  = (Nn * Hh + B - 1) / B;       // 25000
    const int gNode  = (Nn + B - 1) / B;            // 196
    const int gEdge  = (EP + B - 1) / B;            // 2696
    const int gScat  = (EP * 32 + B - 1) / B;       // 86250
    const int gGemm  = ((Nn / 16) + 7) / 8;         // 391 blocks of 8 waves
    const int gW     = (Hh * Hh + B - 1) / B;       // 64

    // buffer rotation: layer l reads `in`, GEMM writes hbuf, aggregation writes obuf;
    // obuf differs from hbuf; for l > 0 hbuf == in (GEMM runs in place, each wave loads its rows first).
    float* bufs[3] = { buf0, buf1, buf2 };
    const float* in = x;

    for (int l = 0; l < 3; ++l) {
        float* hbuf = bufs[l % 3];
        float* obuf = bufs[(l + 1) % 3];
        cvtW_kernel<<<gW, B, 0, stream>>>(Wc + (size_t)l * Hh * Hh, wt16);
        gemm_wmma_kernel<<<gGemm, B, 0, stream>>>(in, wt16, hbuf);
        node_attn_kernel<<<gNode, B, 0, stream>>>(hbuf, a_src + l * Hh, a_dst + l * Hh, es, ed);
        init_node_kernel<<<gNode, B, 0, stream>>>(mInt, z);
        hipMemsetAsync(obuf, 0, (size_t)Nn * Hh * 4, stream);
        edge_max_kernel<<<gEdge, B, 0, stream>>>(esrc, edst, es, ed, ew, mInt);
        edge_exp_kernel<<<gEdge, B, 0, stream>>>(edst, ew, mInt, z);
        edge_scatter_kernel<<<gScat, B, 0, stream>>>(esrc, edst, ew, z, hbuf, obuf);
        bias_relu_kernel<<<gElem, B, 0, stream>>>(obuf, bc + l * Hh);
        in = obuf;
    }

    init_pool_kernel<<<(Gg * Hh + B - 1) / B, B, 0, stream>>>(gInt);
    pool_max_kernel<<<gElem, B, 0, stream>>>(in, batch, gInt);
    mlp_kernel<<<Gg, Hh, 0, stream>>>(gInt, W1, b1, W2, b2, (float*)d_out);
}
